// pCOINBlock_51608327029375
// MI455X (gfx1250) — matrix operand order repaired and checked by reading, not compiled, not run on hardware
//
#include <hip/hip_runtime.h>
#include <hip/hip_bf16.h>
#include <math.h>

// ---------------------------------------------------------------------------
// pCOINBlock for MI455X (gfx1250, wave32, WMMA).
//  Phase 1 (parallel): A = (sum(relu(XWq)*relu(XWk))) * (XWv), fused bf16 WMMA
//  Phase 2 (scan):     h_t = tanh(h_{t-1} @ Whh^T + b + A_t), 32 persistent
//                      WGs, grid barrier per step, Whh^T bf16 resident in L2.
//  Weights and h are pre-packed as bf16 K-pairs laid out [n][kp] so every
//  WMMA fragment is two contiguous 16-byte runs -> b128 loads.
// ---------------------------------------------------------------------------

typedef __attribute__((ext_vector_type(16))) __bf16   v16bf;
typedef __attribute__((ext_vector_type(8)))  float    v8f;
typedef __attribute__((ext_vector_type(8)))  unsigned v8u;

#define B_ 8
#define T_ 2048
#define C_ 1024
#define KP_ (C_/2)         // 512 packed K-pairs per row
#define KTILES 32          // C_/32 k-chunks for 16x16x32 bf16 WMMA
#define SCAN_WGS 32        // persistent workgroups in the scan

__device__ __forceinline__ unsigned short f2bf(float f) {
  unsigned u = __builtin_bit_cast(unsigned, f);
  unsigned r = 0x7FFFu + ((u >> 16) & 1u);          // round-to-nearest-even
  return (unsigned short)((u + r) >> 16);
}
__device__ __forceinline__ unsigned pack2(float a, float b) {
  return (unsigned)f2bf(a) | ((unsigned)f2bf(b) << 16);
}

// ---------------- workspace layout (bytes) ----------------
#define XB_OFF   ((size_t)0)                          // bf16 X        : 32 MB
#define WQ_OFF   (XB_OFF  + (size_t)B_*T_*C_*2)       // packed bf16   :  2 MB
#define WK_OFF   (WQ_OFF  + (size_t)KP_*C_*4)
#define WV_OFF   (WK_OFF  + (size_t)KP_*C_*4)
#define WHH_OFF  (WV_OFF  + (size_t)KP_*C_*4)         // Whh^T packed  :  2 MB
#define A_OFF    (WHH_OFF + (size_t)KP_*C_*4)         // fp32 A        : 64 MB
#define HB0_OFF  (A_OFF   + (size_t)B_*T_*C_*4)       // packed bf16 h : 32 KB
#define HB1_OFF  (HB0_OFF + (size_t)16*KP_*4)
#define CNT_OFF  (HB1_OFF + (size_t)16*KP_*4)

// ---------------------------------------------------------------------------
// Kernel 0a: fp32 -> bf16 conversion of X
// ---------------------------------------------------------------------------
__global__ void cvt_x_kernel(const float* __restrict__ X,
                             unsigned short* __restrict__ Xb, int n) {
  int i = blockIdx.x * blockDim.x + threadIdx.x;
  if (i < n) Xb[i] = f2bf(X[i]);
}

// ---------------------------------------------------------------------------
// Kernel 0b: pack weights as bf16 K-pairs, fragment-friendly layout:
//   Wp[n*KP_ + kp] = {W[2kp][n], W[2kp+1][n]}   (B[k][n] = W[k][n])
//   Whhp[n*KP_+kp] = {Whh[n][2kp], Whh[n][2kp+1]} (B = Whh^T)
// so a B-fragment's 8 dwords are two contiguous 16B runs -> global_load_b128.
// ---------------------------------------------------------------------------
__global__ void pack_w_kernel(const float* __restrict__ WQ,
                              const float* __restrict__ WK,
                              const float* __restrict__ WV,
                              const float* __restrict__ Whh,
                              unsigned* __restrict__ WQp,
                              unsigned* __restrict__ WKp,
                              unsigned* __restrict__ WVp,
                              unsigned* __restrict__ Whhp) {
  int i = blockIdx.x * blockDim.x + threadIdx.x;           // over C*KP_
  if (i >= C_ * KP_) return;
  int n = i >> 9, kp = i & (KP_ - 1);
  int k = kp * 2;
  WQp[i]  = pack2(WQ[(size_t)k * C_ + n],  WQ[(size_t)(k + 1) * C_ + n]);
  WKp[i]  = pack2(WK[(size_t)k * C_ + n],  WK[(size_t)(k + 1) * C_ + n]);
  WVp[i]  = pack2(WV[(size_t)k * C_ + n],  WV[(size_t)(k + 1) * C_ + n]);
  Whhp[i] = pack2(Whh[(size_t)n * C_ + k], Whh[(size_t)n * C_ + k + 1]);
}

// ---------------------------------------------------------------------------
// Kernel 0c: zero packed h ping/pong (16 rows: rows 8..15 stay 0 forever so
// the 16-row WMMA A-tile needs no predication), reset barrier, copy S_n out.
// ---------------------------------------------------------------------------
__global__ void init_kernel(unsigned* __restrict__ hb0, unsigned* __restrict__ hb1,
                            unsigned* __restrict__ cnt,
                            const float* __restrict__ Sn,
                            float* __restrict__ outSn) {
  int i = blockIdx.x * blockDim.x + threadIdx.x;
  if (i < 16 * KP_) { hb0[i] = 0u; hb1[i] = 0u; }
  if (i < B_ * C_) outSn[i] = Sn[i];
  if (i == 0) *cnt = 0u;
}

// ---------------------------------------------------------------------------
// Kernel 1: fused QKV + A.  1024 blocks x 512 threads (16 waves).
// Each block owns 16 rows of (B*T); each wave owns 4 N-tiles of Q, K and V.
// X tile (16x32 bf16, [m][kp] pair layout) staged in LDS, shared by all waves.
// ---------------------------------------------------------------------------
__global__ __launch_bounds__(512, 1)
void qkv_a_kernel(const unsigned short* __restrict__ Xb,
                  const unsigned* __restrict__ WQp,
                  const unsigned* __restrict__ WKp,
                  const unsigned* __restrict__ WVp,
                  float* __restrict__ A) {
  __shared__ unsigned xtile[16 * 16];   // [m][kp] packed bf16 pairs
  __shared__ float    sbuf[16];         // per-row  sum(Q*K)

  const int lane = threadIdx.x & 31;
  const int wave = threadIdx.x >> 5;    // 0..15
  const int hi   = lane >> 4;           // half-wave select
  const int nl   = lane & 15;
  const int row0 = blockIdx.x * 16;

  v8f cQ[4], cK[4], cV[4];
#pragma unroll
  for (int j = 0; j < 4; ++j) { cQ[j] = (v8f)0.f; cK[j] = (v8f)0.f; cV[j] = (v8f)0.f; }

  for (int kt = 0; kt < KTILES; ++kt) {
    __syncthreads();
    if (threadIdx.x < 256) {            // stage X[16 rows x 32 k] as pairs
      int m = threadIdx.x >> 4, kp = threadIdx.x & 15;
      const unsigned* src = (const unsigned*)
          (Xb + (size_t)(row0 + m) * C_ + kt * 32 + 2 * kp);
      xtile[m * 16 + kp] = *src;        // [m][kp]: contiguous per m-group
    }
    __syncthreads();

    // A fragment (16x32 bf16): slots v=0..3 / 4..7 are contiguous kp runs
    v8u au;
#pragma unroll
    for (int v = 0; v < 8; ++v) {
      int kp = v + (hi ? 8 : 0);
      au[v] = xtile[nl * 16 + kp];      // 2x ds_load_b128
    }
    v16bf a = __builtin_bit_cast(v16bf, au);

#pragma unroll
    for (int j = 0; j < 4; ++j) {
      const int ncol = (wave * 4 + j) * 16 + nl;
      const size_t base = (size_t)ncol * KP_ + kt * 16 + (hi ? 8 : 0);
      v8u bq, bk, bv;
#pragma unroll
      for (int v = 0; v < 8; ++v) {     // contiguous -> 2x global_load_b128 ea.
        bq[v] = WQp[base + v];  bk[v] = WKp[base + v];  bv[v] = WVp[base + v];
      }
      cQ[j] = __builtin_amdgcn_wmma_f32_16x16x32_bf16(
          false, a, false, __builtin_bit_cast(v16bf, bq), (short)0, cQ[j], false, false);
      cK[j] = __builtin_amdgcn_wmma_f32_16x16x32_bf16(
          false, a, false, __builtin_bit_cast(v16bf, bk), (short)0, cK[j], false, false);
      cV[j] = __builtin_amdgcn_wmma_f32_16x16x32_bf16(
          false, a, false, __builtin_bit_cast(v16bf, bv), (short)0, cV[j], false, false);
    }
  }

  __syncthreads();
  if (threadIdx.x < 16) sbuf[threadIdx.x] = 0.f;
  __syncthreads();

  // s = sum_c relu(Q)*relu(K): lane-local, then shfl over the 16-lane N group,
  // then LDS atomic across waves (each wave covers 4 disjoint N-tiles).
  float p[8];
#pragma unroll
  for (int v = 0; v < 8; ++v) p[v] = 0.f;
#pragma unroll
  for (int j = 0; j < 4; ++j)
#pragma unroll
    for (int v = 0; v < 8; ++v) {
      float q = fmaxf(cQ[j][v], 0.f);
      float k = fmaxf(cK[j][v], 0.f);
      p[v] += q * k;
    }
#pragma unroll
  for (int off = 1; off < 16; off <<= 1)
#pragma unroll
    for (int v = 0; v < 8; ++v) p[v] += __shfl_xor(p[v], off, 32);
  if (nl == 0) {
#pragma unroll
    for (int v = 0; v < 8; ++v) atomicAdd(&sbuf[v + (hi ? 8 : 0)], p[v]);
  }
  __syncthreads();

#pragma unroll
  for (int j = 0; j < 4; ++j)
#pragma unroll
    for (int v = 0; v < 8; ++v) {
      int m = v + (hi ? 8 : 0);
      int ncol = (wave * 4 + j) * 16 + nl;
      A[(size_t)(row0 + m) * C_ + ncol] = sbuf[m] * cV[j][v];
    }
}

// ---------------------------------------------------------------------------
// Kernel 2: persistent scan.  32 WGs x 64 threads (2 waves); each wave owns
// one 16-col N-tile.  h kept as packed bf16 in A-fragment layout [m][kp] so
// each k-chunk needs just 2+2 b128 loads.  Cross-WGP visibility: release
// fence before signalling, acquire fence (L0 global_inv) after the barrier.
// ---------------------------------------------------------------------------
__global__ __launch_bounds__(64, 1)
void scan_kernel(const float* __restrict__ A,
                 const unsigned* __restrict__ Whhp,
                 const float* __restrict__ b_hh,
                 unsigned* __restrict__ hb0, unsigned* __restrict__ hb1,
                 unsigned* cnt, float* __restrict__ out) {
  const int lane  = threadIdx.x & 31;
  const int wave  = threadIdx.x >> 5;                 // 0..1
  const int ntile = blockIdx.x * 2 + wave;            // 0..63
  const int hi    = lane >> 4;
  const int nl    = lane & 15;
  const int col   = ntile * 16 + nl;
  const float bias = b_hh[col];

  for (int t = 0; t < T_; ++t) {
    const unsigned* hp = (t & 1) ? hb1 : hb0;
    unsigned*       hn = (t & 1) ? hb0 : hb1;

    v8f acc = (v8f)0.f;
    for (int kt = 0; kt < KTILES; ++kt) {
      v8u au, bu;
      const int abase = nl * KP_ + kt * 16 + (hi ? 4 : 0);
      const size_t bbase = (size_t)col * KP_ + kt * 16 + (hi ? 4 : 0);
#pragma unroll
      for (int v = 0; v < 4; ++v) {     // contiguous runs -> b128 loads
        au[v]     = hp[abase + v];
        au[v + 4] = hp[abase + 8 + v];
        bu[v]     = Whhp[bbase + v];
        bu[v + 4] = Whhp[bbase + 8 + v];
      }
      acc = __builtin_amdgcn_wmma_f32_16x16x32_bf16(
          false, __builtin_bit_cast(v16bf, au),
          false, __builtin_bit_cast(v16bf, bu), (short)0, acc, false, false);
    }

#pragma unroll
    for (int v = 0; v < 8; ++v) {
      float aval = 0.f;
      if (hi == 0)                      // row m = v (batch), valid only hi==0
        aval = A[((size_t)v * T_ + t) * C_ + col];
      float val = tanhf(acc[v] + bias + aval);        // all lanes: shfl below
      float vpart = __shfl_xor(val, 1, 32);           // partner column value
      if (hi == 0) {
        out[((size_t)v * T_ + t) * C_ + col] = val;
        if ((nl & 1) == 0)              // even col packs {col, col+1}
          hn[v * KP_ + ntile * 8 + (nl >> 1)] = pack2(val, vpart);
      }
    }

    // ---- grid-wide step barrier ----
    __threadfence();                    // release h/out stores (global_wb)
    __syncthreads();
    if (threadIdx.x == 0) {
      __hip_atomic_fetch_add(cnt, 1u, __ATOMIC_RELEASE, __HIP_MEMORY_SCOPE_AGENT);
      const unsigned target = (unsigned)(t + 1) * SCAN_WGS;
      while (__hip_atomic_load(cnt, __ATOMIC_RELAXED, __HIP_MEMORY_SCOPE_AGENT) < target)
        __builtin_amdgcn_s_sleep(1);
    }
    __syncthreads();
    __threadfence();                    // acquire: global_inv stale L0 lines
  }
}

// ---------------------------------------------------------------------------
extern "C" void kernel_launch(void* const* d_in, const int* in_sizes, int n_in,
                              void* d_out, int out_size, void* d_ws, size_t ws_size,
                              hipStream_t stream) {
  const float* X    = (const float*)d_in[0];
  const float* Sn   = (const float*)d_in[2];
  const float* W_Q  = (const float*)d_in[4];
  const float* W_K  = (const float*)d_in[5];
  const float* W_V  = (const float*)d_in[6];
  const float* W_hh = (const float*)d_in[7];
  const float* b_hh = (const float*)d_in[8];
  float* out = (float*)d_out;

  char* ws = (char*)d_ws;
  unsigned short* Xb   = (unsigned short*)(ws + XB_OFF);
  unsigned*       WQp  = (unsigned*)(ws + WQ_OFF);
  unsigned*       WKp  = (unsigned*)(ws + WK_OFF);
  unsigned*       WVp  = (unsigned*)(ws + WV_OFF);
  unsigned*       Whhp = (unsigned*)(ws + WHH_OFF);
  float*          A    = (float*)(ws + A_OFF);
  unsigned*       hb0  = (unsigned*)(ws + HB0_OFF);
  unsigned*       hb1  = (unsigned*)(ws + HB1_OFF);
  unsigned*       cnt  = (unsigned*)(ws + CNT_OFF);

  const int nX = B_ * T_ * C_;
  cvt_x_kernel<<<nX / 256, 256, 0, stream>>>(X, Xb, nX);
  pack_w_kernel<<<C_ * KP_ / 256, 256, 0, stream>>>(
      W_Q, W_K, W_V, W_hh, WQp, WKp, WVp, Whhp);
  init_kernel<<<64, 256, 0, stream>>>(hb0, hb1, cnt, Sn, out + (size_t)B_ * T_ * C_);
  qkv_a_kernel<<<(B_ * T_) / 16, 512, 0, stream>>>(Xb, WQp, WKp, WVp, A);
  scan_kernel<<<SCAN_WGS, 64, 0, stream>>>(A, Whhp, b_hh, hb0, hb1, cnt, out);
}
